// MHSAModule_16578573762597
// MI455X (gfx1250) — compile-verified, matrix operand order checked by reading, not run on hardware
//
#include <hip/hip_runtime.h>
#include <hip/hip_bf16.h>
#include <cstdint>

typedef __attribute__((ext_vector_type(16))) _Float16 v16h;
typedef __attribute__((ext_vector_type(8)))  float    v8f;

#define TSEQ   1024
#define DMODEL 512
#define NH     8
#define DHEAD  64
#define BTOT   8192   // B*T

// ---------------------------------------------------------------------------
// Fragment loader for v_wmma_f32_16x16x32_f16 (A: 16xK rows, B: KxN cols).
// ISA 7.12.2 16-bit A layout: lane&15 selects the M (or N) index; lanes<16
// hold K = {k0..k0+7, k0+16..k0+23}, lanes>=16 hold K = {k0+8..k0+15,
// k0+24..k0+31}; each VGPR packs a contiguous K pair.  `base` must point at a
// matrix whose K dimension is contiguous with row stride `stride` elements.
// ---------------------------------------------------------------------------
__device__ __forceinline__ v16h load_frag(const _Float16* __restrict__ base,
                                          int stride, int rc, int k0, int lane) {
  union { uint32_t u[8]; v16h h; } f;
  const _Float16* p = base + (size_t)rc * stride;
  const int klo = k0 + ((lane & 16) ? 8 : 0);
  const int khi = klo + 16;
#pragma unroll
  for (int j = 0; j < 4; ++j)
    f.u[j] = *(const uint32_t*)(p + klo + 2 * j);
#pragma unroll
  for (int j = 0; j < 4; ++j)
    f.u[4 + j] = *(const uint32_t*)(p + khi + 2 * j);
  return f.h;
}

__device__ __forceinline__ v8f wmma_f16(v16h a, v16h b, v8f c) {
  return __builtin_amdgcn_wmma_f32_16x16x32_f16(false, a, false, b, (short)0, c,
                                                false, false);
}

// ---------------------------------------------------------------------------
// Kernel 1: fused LayerNorm + cast to f16.  One block per (b,t) row.
// ---------------------------------------------------------------------------
__global__ __launch_bounds__(128) void ln_cast_kernel(
    const float* __restrict__ in, const float* __restrict__ gamma,
    const float* __restrict__ beta, _Float16* __restrict__ xh) {
  const int row = blockIdx.x;
  const int tid = threadIdx.x;
  const float4 v = ((const float4*)(in + (size_t)row * DMODEL))[tid];
  float s  = v.x + v.y + v.z + v.w;
  float sq = v.x * v.x + v.y * v.y + v.z * v.z + v.w * v.w;
#pragma unroll
  for (int off = 16; off; off >>= 1) {
    s  += __shfl_xor(s, off, 32);
    sq += __shfl_xor(sq, off, 32);
  }
  __shared__ float red[2][4];
  if ((tid & 31) == 0) { red[0][tid >> 5] = s; red[1][tid >> 5] = sq; }
  __syncthreads();
  s  = red[0][0] + red[0][1] + red[0][2] + red[0][3];
  sq = red[1][0] + red[1][1] + red[1][2] + red[1][3];
  const float mean = s * (1.0f / DMODEL);
  const float var  = sq * (1.0f / DMODEL) - mean * mean;
  const float rs   = rsqrtf(var + 1e-3f);
  const int c = tid * 4;
  _Float16* o = xh + (size_t)row * DMODEL + c;
  o[0] = (_Float16)((v.x - mean) * rs * gamma[c + 0] + beta[c + 0]);
  o[1] = (_Float16)((v.y - mean) * rs * gamma[c + 1] + beta[c + 1]);
  o[2] = (_Float16)((v.z - mean) * rs * gamma[c + 2] + beta[c + 2]);
  o[3] = (_Float16)((v.w - mean) * rs * gamma[c + 3] + beta[c + 3]);
}

// Kernel 2: plain f32 -> f16 cast (positional encodings).
__global__ void cast_f2h_kernel(const float* __restrict__ in,
                                _Float16* __restrict__ out, int n) {
  const int i = blockIdx.x * blockDim.x + threadIdx.x;
  if (i < n) out[i] = (_Float16)in[i];
}

// ---------------------------------------------------------------------------
// Kernel 3: repack weights to f16, K-contiguous for WMMA B operands.
//  y<4 : [H,D,DH] -> Wt[h*64+o][i]      (for x @ W per-head projections)
//  y=4 : proj_k [H,DH,D] -> Wt[d][h*64+o] (for O @ proj)
// ---------------------------------------------------------------------------
__global__ __launch_bounds__(256) void repack_kernel(
    const float* __restrict__ qk, const float* __restrict__ kk,
    const float* __restrict__ vk, const float* __restrict__ pk,
    const float* __restrict__ projk,
    _Float16* __restrict__ wq, _Float16* __restrict__ wk,
    _Float16* __restrict__ wv, _Float16* __restrict__ wp,
    _Float16* __restrict__ wproj) {
  const int y = blockIdx.y;
  const int idx = blockIdx.x * 256 + threadIdx.x;   // [0, 512*512)
  if (y < 4) {
    const float* W = (y == 0) ? qk : (y == 1) ? kk : (y == 2) ? vk : pk;
    _Float16*    O = (y == 0) ? wq : (y == 1) ? wk : (y == 2) ? wv : wp;
    const int n = idx >> 9, i = idx & 511;
    const int h = n >> 6,  o = n & 63;
    O[idx] = (_Float16)W[((size_t)h * DMODEL + i) * DHEAD + o];
  } else {
    const int d = idx >> 9, n = idx & 511;
    wproj[idx] = (_Float16)projk[(size_t)n * DMODEL + d];
  }
}

// ---------------------------------------------------------------------------
// Kernel 4: QKV/P projections.  One wave computes a 16x64 tile (K=512).
//  z=0: q  -> writes qu = q+qb+pos_bias_u and qv = q+qb+pos_bias_v  [bt][512]
//  z=1: k  -> k_h  [b,h,t,dh]   (K-contiguous B operand for q.k^T)
//  z=2: v  -> vT_h [b,h,dh,t]   (K-contiguous B operand for probs.v)
//  z=3: p  -> p_h  [b,h,t,dh]
// ---------------------------------------------------------------------------
__global__ __launch_bounds__(32) void qkvp_kernel(
    const _Float16* __restrict__ xh, const _Float16* __restrict__ ph,
    const _Float16* __restrict__ wq, const _Float16* __restrict__ wk,
    const _Float16* __restrict__ wv, const _Float16* __restrict__ wp,
    const float* __restrict__ qb, const float* __restrict__ kb,
    const float* __restrict__ vb, const float* __restrict__ pbu,
    const float* __restrict__ pbv,
    _Float16* __restrict__ qu, _Float16* __restrict__ qv,
    _Float16* __restrict__ kout, _Float16* __restrict__ pout,
    _Float16* __restrict__ vtout) {
  const int z = blockIdx.z;
  const _Float16* A = (z == 3) ? ph : xh;
  const _Float16* W = (z == 0) ? wq : (z == 1) ? wk : (z == 2) ? wv : wp;
  const int lane = threadIdx.x;
  const int m0 = blockIdx.x * 16;
  const int n0 = blockIdx.y * 64;
  const int rowA = m0 + (lane & 15);

  v8f acc[4] = {{}, {}, {}, {}};
  for (int k0 = 0; k0 < DMODEL; k0 += 32) {
    __builtin_prefetch(A + (size_t)rowA * DMODEL + k0 + 64, 0, 1);
    v16h a = load_frag(A, DMODEL, rowA, k0, lane);
#pragma unroll
    for (int s = 0; s < 4; ++s) {
      v16h bf = load_frag(W, DMODEL, n0 + s * 16 + (lane & 15), k0, lane);
      acc[s] = wmma_f16(a, bf, acc[s]);
    }
  }
  const int Mb = (lane & 16) ? 8 : 0;
  const int N  = lane & 15;
#pragma unroll
  for (int s = 0; s < 4; ++s) {
#pragma unroll
    for (int r = 0; r < 8; ++r) {
      const int bt  = m0 + Mb + r;
      const int col = n0 + s * 16 + N;
      const int hh = col >> 6, oo = col & 63;
      const int bb = bt >> 10, tt = bt & 1023;
      float cv = acc[s][r];
      if (z == 0) {
        cv += qb[col];
        qu[(size_t)bt * DMODEL + col] = (_Float16)(cv + pbu[col]);
        qv[(size_t)bt * DMODEL + col] = (_Float16)(cv + pbv[col]);
      } else if (z == 1) {
        kout[(((size_t)bb * NH + hh) * TSEQ + tt) * DHEAD + oo] =
            (_Float16)(cv + kb[col]);
      } else if (z == 2) {
        vtout[(((size_t)bb * NH + hh) * DHEAD + oo) * TSEQ + tt] =
            (_Float16)(cv + vb[col]);
      } else {
        pout[(((size_t)bb * NH + hh) * TSEQ + tt) * DHEAD + oo] = (_Float16)cv;
      }
    }
  }
}

// ---------------------------------------------------------------------------
// Kernel 5: fused rel-position attention per (b, h, 32-row tile).
// rel_shift scatter: P[r,c] -> (r, c+r-T+1) if c >= T-1-r, else (r-1, c+r+1);
// shifted (n, n+1) is the zero-pad slot (never written). Targets are unique.
// ---------------------------------------------------------------------------
#define SP  1028          // f32 LDS row stride (padded)
#define PRP 1032          // f16 LDS row stride (padded)
#define ATTN_LDS (32 * SP * 4 + 32 * PRP * 2 + 32 * 4)

__global__ __launch_bounds__(256) void attn_kernel(
    const _Float16* __restrict__ qu, const _Float16* __restrict__ qv,
    const _Float16* __restrict__ kmat, const _Float16* __restrict__ pmat,
    const _Float16* __restrict__ vt, _Float16* __restrict__ outh) {
  extern __shared__ char smem[];
  float*    S    = (float*)smem;                         // 32 x SP logits
  _Float16* Pr   = (_Float16*)(smem + 32 * SP * 4);      // 32 x PRP probs
  float*    rsum = (float*)(smem + 32 * SP * 4 + 32 * PRP * 2);

  const int lane = threadIdx.x & 31;
  const int w    = threadIdx.x >> 5;
  const int n0 = blockIdx.x * 32;
  const int h  = blockIdx.y;
  const int b  = blockIdx.z;
  const int bh = b * NH + h;

  const _Float16* QU = qu + ((size_t)b * TSEQ + n0) * DMODEL + h * DHEAD;
  const _Float16* QV = qv + ((size_t)b * TSEQ + n0) * DMODEL + h * DHEAD;
  const _Float16* K  = kmat + (size_t)bh * TSEQ * DHEAD;   // [t][dh]
  const _Float16* P  = pmat + (size_t)bh * TSEQ * DHEAD;   // [t][dh]
  const _Float16* VT = vt + (size_t)bh * DHEAD * TSEQ;     // [dh][t]

  // ---- phase 1: content logits U = (q+u) . k^T  ->  S ----
  for (int idx = w; idx < 128; idx += 8) {            // 2 row-tiles x 64 col-tiles
    const int rt = idx >> 6, ct = idx & 63;
    v8f acc = {};
#pragma unroll
    for (int k0 = 0; k0 < DHEAD; k0 += 32) {
      v16h a  = load_frag(QU, DMODEL, rt * 16 + (lane & 15), k0, lane);
      v16h bf = load_frag(K, DHEAD, ct * 16 + (lane & 15), k0, lane);
      acc = wmma_f16(a, bf, acc);
    }
    const int Mb = (lane & 16) ? 8 : 0;
    const int N  = lane & 15;
#pragma unroll
    for (int r = 0; r < 8; ++r)
      S[(rt * 16 + Mb + r) * SP + ct * 16 + N] = acc[r];
  }
  __syncthreads();

  // ---- phase 2: position logits, rel-shift scatter-add into S ----
  for (int idx = w; idx < 192; idx += 8) {            // 3 row-tiles (rows n0..n0+47)
    const int rt = idx >> 6, ct = idx & 63;
    int rl = rt * 16 + (lane & 15);
    if (n0 + rl > TSEQ - 1) rl = TSEQ - 1 - n0;       // clamp load only
    v8f acc = {};
#pragma unroll
    for (int k0 = 0; k0 < DHEAD; k0 += 32) {
      v16h a  = load_frag(QV, DMODEL, rl, k0, lane);
      v16h bf = load_frag(P, DHEAD, ct * 16 + (lane & 15), k0, lane);
      acc = wmma_f16(a, bf, acc);
    }
    const int Mb = (lane & 16) ? 8 : 0;
    const int N  = lane & 15;
#pragma unroll
    for (int r = 0; r < 8; ++r) {
      const int rg = n0 + rt * 16 + Mb + r;           // pre-shift global row
      if (rg >= TSEQ) continue;
      const int cg = ct * 16 + N;                     // pre-shift column
      int n, m;
      if (cg >= TSEQ - 1 - rg) { n = rg;     m = cg + rg - (TSEQ - 1); }
      else                     { n = rg - 1; m = cg + rg + 1;          }
      if (n >= n0 && n < n0 + 32) S[(n - n0) * SP + m] += acc[r];
    }
  }
  __syncthreads();

  // ---- phase 3: row softmax (unnormalized exp -> f16 probs, sums to LDS) ----
#pragma unroll
  for (int rr = 0; rr < 4; ++rr) {
    const int row = w * 4 + rr;
    float vv[32];
    float mx = -3.4e38f;
#pragma unroll
    for (int j = 0; j < 32; ++j) {
      vv[j] = S[row * SP + j * 32 + lane] * 0.125f;   // 1/sqrt(DH)
      mx = fmaxf(mx, vv[j]);
    }
#pragma unroll
    for (int off = 16; off; off >>= 1) mx = fmaxf(mx, __shfl_xor(mx, off, 32));
    float sm = 0.f;
#pragma unroll
    for (int j = 0; j < 32; ++j) {
      const float e = __expf(vv[j] - mx);
      Pr[row * PRP + j * 32 + lane] = (_Float16)e;
      sm += e;
    }
#pragma unroll
    for (int off = 16; off; off >>= 1) sm += __shfl_xor(sm, off, 32);
    if (lane == 0) rsum[row] = sm;
  }
  __syncthreads();

  // ---- phase 4: O = probs @ v  (A fragments straight from LDS) ----
  {
    const int rt = w >> 2;
    const int o0 = (w & 3) * 16;
    const int arow = rt * 16 + (lane & 15);
    v8f acc = {};
    for (int k0 = 0; k0 < TSEQ; k0 += 32) {
      union { uint32_t u[8]; v16h h; } fa;
      const int klo = k0 + ((lane & 16) ? 8 : 0);
      const int khi = klo + 16;
#pragma unroll
      for (int j = 0; j < 4; ++j)
        fa.u[j] = *(const uint32_t*)&Pr[arow * PRP + klo + 2 * j];
#pragma unroll
      for (int j = 0; j < 4; ++j)
        fa.u[4 + j] = *(const uint32_t*)&Pr[arow * PRP + khi + 2 * j];
      v16h bf = load_frag(VT, TSEQ, o0 + (lane & 15), k0, lane);
      acc = wmma_f16(fa.h, bf, acc);
    }
    const int Mb = (lane & 16) ? 8 : 0;
    const int N  = lane & 15;
#pragma unroll
    for (int r = 0; r < 8; ++r) {
      const int row = rt * 16 + Mb + r;
      const float inv = 1.0f / rsum[row];
      outh[((size_t)b * TSEQ + n0 + row) * DMODEL + h * DHEAD + o0 + N] =
          (_Float16)(acc[r] * inv);
    }
  }
}

// ---------------------------------------------------------------------------
// Kernel 6: output projection + bias + residual -> f32.
// ---------------------------------------------------------------------------
__global__ __launch_bounds__(32) void proj_kernel(
    const _Float16* __restrict__ oh, const _Float16* __restrict__ wproj,
    const float* __restrict__ projb, const float* __restrict__ resid,
    float* __restrict__ out) {
  const int lane = threadIdx.x;
  const int m0 = blockIdx.x * 16;
  const int n0 = blockIdx.y * 64;
  const int rowA = m0 + (lane & 15);
  v8f acc[4] = {{}, {}, {}, {}};
  for (int k0 = 0; k0 < DMODEL; k0 += 32) {
    __builtin_prefetch(oh + (size_t)rowA * DMODEL + k0 + 64, 0, 1);
    v16h a = load_frag(oh, DMODEL, rowA, k0, lane);
#pragma unroll
    for (int s = 0; s < 4; ++s) {
      v16h bf = load_frag(wproj, DMODEL, n0 + s * 16 + (lane & 15), k0, lane);
      acc[s] = wmma_f16(a, bf, acc[s]);
    }
  }
  const int Mb = (lane & 16) ? 8 : 0;
  const int N  = lane & 15;
#pragma unroll
  for (int s = 0; s < 4; ++s)
#pragma unroll
    for (int r = 0; r < 8; ++r) {
      const int bt = m0 + Mb + r;
      const int d  = n0 + s * 16 + N;
      out[(size_t)bt * DMODEL + d] =
          resid[(size_t)bt * DMODEL + d] + projb[d] + acc[s][r];
    }
}

// ---------------------------------------------------------------------------
extern "C" void kernel_launch(void* const* d_in, const int* in_sizes, int n_in,
                              void* d_out, int out_size, void* d_ws, size_t ws_size,
                              hipStream_t stream) {
  (void)in_sizes; (void)n_in; (void)out_size; (void)ws_size;
  const float* inputs = (const float*)d_in[0];
  const float* pos    = (const float*)d_in[1];
  const float* ln_g   = (const float*)d_in[2];
  const float* ln_b   = (const float*)d_in[3];
  const float* qk     = (const float*)d_in[4];
  const float* qb     = (const float*)d_in[5];
  const float* kk     = (const float*)d_in[6];
  const float* kb     = (const float*)d_in[7];
  const float* vk     = (const float*)d_in[8];
  const float* vb     = (const float*)d_in[9];
  const float* pk     = (const float*)d_in[10];
  const float* pbu    = (const float*)d_in[11];
  const float* pbv    = (const float*)d_in[12];
  const float* projk  = (const float*)d_in[13];
  const float* projb  = (const float*)d_in[14];
  float* out = (float*)d_out;

  char* ws = (char*)d_ws;
  size_t off = 0;
  auto take = [&](size_t bytes) {
    char* p = ws + off;
    off += (bytes + 255) & ~(size_t)255;
    return p;
  };
  const size_t ACT = (size_t)BTOT * DMODEL * sizeof(_Float16);   // 8 MB
  const size_t WT  = (size_t)DMODEL * DMODEL * sizeof(_Float16); // 512 KB
  _Float16* xh    = (_Float16*)take(ACT);
  _Float16* ph    = (_Float16*)take(ACT);
  _Float16* quh   = (_Float16*)take(ACT);
  _Float16* qvh   = (_Float16*)take(ACT);
  _Float16* kh    = (_Float16*)take(ACT);
  _Float16* pph   = (_Float16*)take(ACT);
  _Float16* vth   = (_Float16*)take(ACT);
  _Float16* ohh   = (_Float16*)take(ACT);
  _Float16* wq    = (_Float16*)take(WT);
  _Float16* wk    = (_Float16*)take(WT);
  _Float16* wv    = (_Float16*)take(WT);
  _Float16* wp    = (_Float16*)take(WT);
  _Float16* wproj = (_Float16*)take(WT);

  ln_cast_kernel<<<BTOT, 128, 0, stream>>>(inputs, ln_g, ln_b, xh);
  cast_f2h_kernel<<<(BTOT * DMODEL + 255) / 256, 256, 0, stream>>>(
      pos, ph, BTOT * DMODEL);
  repack_kernel<<<dim3(1024, 5), 256, 0, stream>>>(qk, kk, vk, pk, projk,
                                                   wq, wk, wv, wp, wproj);
  qkvp_kernel<<<dim3(BTOT / 16, DMODEL / 64, 4), 32, 0, stream>>>(
      xh, ph, wq, wk, wv, wp, qb, kb, vb, pbu, pbv, quh, qvh, kh, pph, vth);
  (void)hipFuncSetAttribute((const void*)attn_kernel,
                            hipFuncAttributeMaxDynamicSharedMemorySize,
                            ATTN_LDS);
  attn_kernel<<<dim3(TSEQ / 32, NH, 8), 256, ATTN_LDS, stream>>>(
      quh, qvh, kh, pph, vth, ohh);
  proj_kernel<<<dim3(BTOT / 16, DMODEL / 64), 32, 0, stream>>>(
      ohh, wproj, projb, inputs, out);
}
